// TransposeDISCOBlock_27582279975307
// MI455X (gfx1250) — compile-verified, matrix operand order checked by reading, not run on hardware
//
#include <hip/hip_runtime.h>
#include <hip/hip_bf16.h>

// ---------- problem constants ----------
#define NB     4      // batch
#define CIN    64
#define COUT   64
#define HI     128
#define WI     128
#define HO     256
#define WO     256
#define RAD    4      // Ry = Rx = 4
#define KS     9      // 9x9 dense kernel
#define KBAS   9      // basis count
#define UPS    72     // padded channel stride per up-patch position (144B: 16B aligned, bank-clean)

#define FILT_ELEMS (81 * 2 * 4 * 32 * 16)   // tap * kblock * cg * lane * 16 = 331776 bf16
#define FILT_BYTES (FILT_ELEMS * 2)

typedef __attribute__((ext_vector_type(16))) __bf16       v16bf;
typedef __attribute__((ext_vector_type(8)))  float        v8f;
typedef __attribute__((ext_vector_type(4)))  unsigned int uint4v;

union BV { uint4v q[2]; v16bf v; };

// CDNA5 async global->LDS copy path (ASYNCcnt-tracked, bypasses VGPRs).
// Guarded so the file still compiles cleanly on toolchains lacking the builtins.
#if defined(__gfx1250__) && __has_builtin(__builtin_amdgcn_global_load_async_to_lds_b32) && __has_builtin(__builtin_amdgcn_s_wait_asynccnt)
#define HAVE_ASYNC_LDS 1
#define AS1 __attribute__((address_space(1)))
#define AS3 __attribute__((address_space(3)))
#endif

// ---------- DISCO basis, evaluated on device (all constants) ----------
__device__ __forceinline__ float disco_psi(int b, int ty, int tx) {
    const float h    = 1.0f / 256.0f;         // hx = hy
    const float rcut = 0.015f;                 // RADIUS_CUTOFF / 2
    const float dr   = rcut * 0.5f;            // rcut / (nr-1), nr = 3
    const float TWO_PI = 6.28318530717958647692f;
    const float dphi   = TWO_PI * 0.25f;       // 2*pi / nphi, nphi = 4
    float dy = (float)(ty - RAD) * h;
    float dx = (float)(tx - RAD) * h;
    float r  = sqrtf(dy * dy + dx * dx);
    float val;
    if (b == 0) {
        val = (r <= dr && r <= rcut) ? (1.0f - r / dr) : 0.0f;
    } else {
        int   ir = (b - 1) / 4 + 1;
        int   ip = (b - 1) % 4;
        float c  = (float)ir * dr;
        float ad = fabsf(r - c);
        float rv = (ad <= dr && r <= rcut) ? (1.0f - ad / dr) : 0.0f;
        float phi = atan2f(dy, dx);
        if (phi < 0.0f) phi += TWO_PI;
        float da = fabsf(phi - (float)ip * dphi);
        da = fminf(da, TWO_PI - da);
        float pv = (da <= dphi) ? (1.0f - da / dphi) : 0.0f;
        val = rv * pv;
    }
    return val * (h * h);  // quadrature weight
}

// ---------- kernel 0: zero the stats buffer ----------
__global__ void zero_stats(float* __restrict__ g) {
    int t = threadIdx.x;         // 256 threads, 512 floats
    g[t]        = 0.0f;
    g[t + 256]  = 0.0f;
}

// ---------- kernel 1: weight x psi -> dense filter in WMMA-B lane layout (bf16) ----------
// layout: fB[ ((tap*2 + kb)*4 + cg)*512 + lane*16 + j ]
// lane<16:  n = lane,      j<8 -> k = j,     j>=8 -> k = j+8
// lane>=16: n = lane-16,   j<8 -> k = j+8,   j>=8 -> k = j+16
__global__ void build_filter(const float* __restrict__ w, __bf16* __restrict__ fB) {
    int idx = blockIdx.x * 256 + threadIdx.x;
    if (idx >= FILT_ELEMS) return;
    int j    = idx & 15;
    int lane = (idx >> 4) & 31;
    int cg   = (idx >> 9) & 3;
    int kb   = (idx >> 11) & 1;
    int tap  = idx >> 12;            // 0..80
    int ty = tap / 9, tx = tap - ty * 9;
    int k32  = (lane < 16) ? ((j < 8) ? j : j + 8)
                           : ((j < 8) ? j + 8 : j + 16);
    int cin  = kb * 32 + k32;
    int och  = cg * 16 + (lane & 15);
    const float* wp = w + (och * CIN + cin) * KBAS;
    float acc = 0.0f;
#pragma unroll
    for (int b = 0; b < KBAS; ++b)
        acc += wp[b] * disco_psi(b, ty, tx);
    fB[idx] = (__bf16)acc;
}

// ---------- kernel 2: fused upsample + implicit-GEMM conv (WMMA) + stats ----------
// Block: 256 threads (8 waves). Tile: 8x8 output pixels x all 64 out-channels.
// Wave w: pixel-group pg = w&3 (16 px), channel-groups cg = 2*(w>>2), +1.
__global__ __launch_bounds__(256) void conv_kernel(const float* __restrict__ img,
                                                   const __bf16* __restrict__ fB,
                                                   float* __restrict__ out,
                                                   float* __restrict__ gstats) {
    __shared__ float  raw[CIN * 10 * 10];        // 25.6 KB
    __shared__ __bf16 up[256 * UPS];             // 36.9 KB (16x16 positions x 72)
    __shared__ float  sst[COUT * 2];             // 0.5 KB

    const int tid = threadIdx.x;
    const int n   = blockIdx.z;
    const int oy0 = blockIdx.y * 8;
    const int ox0 = blockIdx.x * 8;

    if (tid < COUT * 2) sst[tid] = 0.0f;

    const float S = 127.0f / 255.0f;             // align_corners 2x scale
    const int uy0 = oy0 - RAD, ux0 = ox0 - RAD;
    const int uyv = uy0 < 0 ? 0 : uy0;
    const int uxv = ux0 < 0 ? 0 : ux0;
    const int iyB = (int)((float)uyv * S);       // nonneg -> trunc == floor
    const int ixB = (int)((float)uxv * S);

    // ---- stage raw input tile 64 x 10 x 10 ----
#ifdef HAVE_ASYNC_LDS
    // async global->LDS (GLOBAL_LOAD_ASYNC_TO_LDS_B32): per-lane src/dst, no VGPR round-trip
    for (int e = tid; e < CIN * 100; e += 256) {
        int c  = e / 100;
        int r2 = e - c * 100;
        int ly = r2 / 10, lx = r2 - ly * 10;
        int gy = iyB + ly; gy = gy > HI - 1 ? HI - 1 : gy;
        int gx = ixB + lx; gx = gx > WI - 1 ? WI - 1 : gx;
        __builtin_amdgcn_global_load_async_to_lds_b32(
            (AS1 int*)(img + ((n * CIN + c) * HI + gy) * WI + gx),
            (AS3 int*)(raw + e),
            0, 0);
    }
    __builtin_amdgcn_s_wait_asynccnt(0);
#else
    for (int e = tid; e < CIN * 100; e += 256) {
        int c  = e / 100;
        int r2 = e - c * 100;
        int ly = r2 / 10, lx = r2 - ly * 10;
        int gy = iyB + ly; gy = gy > HI - 1 ? HI - 1 : gy;
        int gx = ixB + lx; gx = gx > WI - 1 ? WI - 1 : gx;
        raw[e] = img[((n * CIN + c) * HI + gy) * WI + gx];
    }
#endif
    __syncthreads();

    // ---- build 16x16 upsampled halo patch, bf16, channel-innermost ----
    for (int e = tid; e < 256 * CIN; e += 256) {
        int c   = e & 63;
        int pos = e >> 6;
        int px_ = pos & 15, py_ = pos >> 4;
        int guy = uy0 + py_, gux = ux0 + px_;
        float v = 0.0f;
        if (guy >= 0 && guy < HO && gux >= 0 && gux < WO) {
            float sy = (float)guy * S, sx = (float)gux * S;
            int iy0 = (int)sy, ix0 = (int)sx;
            float wy = sy - (float)iy0, wx = sx - (float)ix0;
            int iy1 = iy0 + 1 > HI - 1 ? HI - 1 : iy0 + 1;
            int ix1 = ix0 + 1 > WI - 1 ? WI - 1 : ix0 + 1;
            int ly0 = iy0 - iyB, ly1 = iy1 - iyB;
            int lx0 = ix0 - ixB, lx1 = ix1 - ixB;
            float v00 = raw[(c * 10 + ly0) * 10 + lx0];
            float v01 = raw[(c * 10 + ly0) * 10 + lx1];
            float v10 = raw[(c * 10 + ly1) * 10 + lx0];
            float v11 = raw[(c * 10 + ly1) * 10 + lx1];
            v = (1.0f - wy) * ((1.0f - wx) * v00 + wx * v01)
              +         wy  * ((1.0f - wx) * v10 + wx * v11);
        }
        up[pos * UPS + c] = (__bf16)v;
    }
    __syncthreads();

    // ---- implicit GEMM over 81 taps x 2 K-blocks, bf16 WMMA, f32 accum ----
    const int lane = tid & 31;
    const int wv   = tid >> 5;
    const int pg   = wv & 3;            // pixel group: rows 2pg, 2pg+1 of 8x8 tile
    const int cg0  = (wv >> 2) * 2;     // out-channel groups cg0, cg0+1
    const int m    = lane & 15;
    const int mrow = 2 * pg + (m >> 3); // pixel row in tile
    const int mcol = m & 7;             // pixel col in tile
    const int coff = (lane < 16) ? 0 : 8;

    v8f acc0 = {};
    v8f acc1 = {};

    for (int ty = 0; ty < KS; ++ty) {
        for (int tx = 0; tx < KS; ++tx) {
            const int tap  = ty * KS + tx;
            const int posb = (mrow + ty) * 16 + (mcol + tx);
#pragma unroll
            for (int kb = 0; kb < 2; ++kb) {
                const int aoff = posb * UPS + kb * 32 + coff;   // 16B-aligned
                BV a;
                a.q[0] = *(const uint4v*)(up + aoff);
                a.q[1] = *(const uint4v*)(up + aoff + 16);
                const __bf16* bbase = fB + ((tap * 2 + kb) * 4) * 512 + lane * 16;
                BV b0, b1;
                { const uint4v* p = (const uint4v*)(bbase + cg0 * 512);       b0.q[0] = p[0]; b0.q[1] = p[1]; }
                { const uint4v* p = (const uint4v*)(bbase + (cg0 + 1) * 512); b1.q[0] = p[0]; b1.q[1] = p[1]; }
                acc0 = __builtin_amdgcn_wmma_f32_16x16x32_bf16(false, a.v, false, b0.v, (short)0, acc0, false, false);
                acc1 = __builtin_amdgcn_wmma_f32_16x16x32_bf16(false, a.v, false, b1.v, (short)0, acc1, false, false);
            }
            if (tap < 80)
                __builtin_prefetch(fB + ((tap + 1) * 2 * 4) * 512 + lane * 16, 0, 3);
        }
    }

    // ---- write y (raw conv) + accumulate per-(n,och) sum / sumsq ----
    const int och0 = cg0 * 16 + m;
    const int py   = 2 * pg + ((lane < 16) ? 0 : 1);
    const int gy   = oy0 + py;
    float* o0 = out + (((size_t)(n * COUT + och0)       * HO + gy) * WO + ox0);
    float* o1 = out + (((size_t)(n * COUT + och0 + 16)  * HO + gy) * WO + ox0);
    float s0 = 0.f, q0 = 0.f, s1 = 0.f, q1 = 0.f;
#pragma unroll
    for (int r = 0; r < 8; ++r) { float v = acc0[r]; o0[r] = v; s0 += v; q0 += v * v; }
#pragma unroll
    for (int r = 0; r < 8; ++r) { float v = acc1[r]; o1[r] = v; s1 += v; q1 += v * v; }

    atomicAdd(&sst[och0 * 2],            s0);
    atomicAdd(&sst[och0 * 2 + 1],        q0);
    atomicAdd(&sst[(och0 + 16) * 2],     s1);
    atomicAdd(&sst[(och0 + 16) * 2 + 1], q1);
    __syncthreads();
    if (tid < COUT * 2)
        atomicAdd(&gstats[n * COUT * 2 + tid], sst[tid]);
}

// ---------- kernel 3: finalize stats -> (mean, rsqrt(var+eps)) ----------
__global__ void finalize_stats(float* __restrict__ g) {
    int t = threadIdx.x;                 // 256 = NB*COUT
    float s = g[t * 2], q = g[t * 2 + 1];
    const float inv_cnt = 1.0f / (float)(HO * WO);
    float mean = s * inv_cnt;
    float var  = q * inv_cnt - mean * mean;
    g[t * 2]     = mean;
    g[t * 2 + 1] = rsqrtf(var + 1e-5f);
}

// ---------- kernel 4: in-place InstanceNorm + LeakyReLU (float4) ----------
__global__ void norm_kernel(float* __restrict__ out, const float* __restrict__ g) {
    int i = blockIdx.x * 256 + threadIdx.x;      // float4 index, 4,194,304 total
    float4 v = ((float4*)out)[i];
    int plane = i >> 14;                          // (n*64+c): 65536 elems / 4 per float4
    float mean = g[plane * 2];
    float inv  = g[plane * 2 + 1];
    float a = (v.x - mean) * inv;
    float b = (v.y - mean) * inv;
    float c = (v.z - mean) * inv;
    float d = (v.w - mean) * inv;
    v.x = a >= 0.f ? a : 0.2f * a;
    v.y = b >= 0.f ? b : 0.2f * b;
    v.z = c >= 0.f ? c : 0.2f * c;
    v.w = d >= 0.f ? d : 0.2f * d;
    ((float4*)out)[i] = v;
}

extern "C" void kernel_launch(void* const* d_in, const int* in_sizes, int n_in,
                              void* d_out, int out_size, void* d_ws, size_t ws_size,
                              hipStream_t stream) {
    const float* img = (const float*)d_in[0];   // [4,64,128,128] f32
    const float* w   = (const float*)d_in[1];   // [64,64,9] f32
    float* out       = (float*)d_out;           // [4,64,256,256] f32

    __bf16* fB     = (__bf16*)d_ws;                         // 331,776 bf16
    float*  gstats = (float*)((char*)d_ws + FILT_BYTES);    // 512 f32

    zero_stats<<<1, 256, 0, stream>>>(gstats);
    build_filter<<<FILT_ELEMS / 256, 256, 0, stream>>>(w, fB);

    dim3 grid(WO / 8, HO / 8, NB);              // (32, 32, 4) -> 4096 blocks
    conv_kernel<<<grid, 256, 0, stream>>>(img, fB, out, gstats);

    finalize_stats<<<1, 256, 0, stream>>>(gstats);

    norm_kernel<<<(NB * COUT * HO * WO / 4) / 256, 256, 0, stream>>>(out, gstats);
}
